// EdgeFtLayerONNX_60301340835934
// MI455X (gfx1250) — matrix operand order repaired and checked by reading, not compiled, not run on hardware
//
#include <hip/hip_runtime.h>

// EdgeFtLayer for MI455X (gfx1250): bf16 WMMA GEMMs, LDS-staged (pre-swizzled)
// weights, scatter-softmax via L2-resident float atomics, bf16 logits spill.
//
// ws layout: segMax f32[N*128] | num f32[N*128] | den f32[N*128] |
//            logits bf16[E*128]  => ~205 MB required in d_ws.
// edge_index treated as int32 per harness convention ("integer -> const int*").

#define NN 50000
#define NE 500000
#define VI 128
#define VO 128
#define EI 64
#define EO 64
#define CAT 320   // 2*VI + EI

typedef __attribute__((ext_vector_type(16))) __bf16    v16bf;
typedef __attribute__((ext_vector_type(8)))  float     v8f;
typedef __attribute__((ext_vector_type(8)))  unsigned  v8u;
typedef __attribute__((ext_vector_type(4)))  unsigned  v4u;
typedef __attribute__((ext_vector_type(4)))  float     v4f;

static __device__ __forceinline__ unsigned short f2bf(float f) {
  unsigned u = __float_as_uint(f);
  u += 0x7FFFu + ((u >> 16) & 1u);        // round-to-nearest-even
  return (unsigned short)(u >> 16);
}
static __device__ __forceinline__ float bf2f(unsigned short h) {
  return __uint_as_float((unsigned)h << 16);
}

static __device__ __forceinline__ v8f wmma_bf16(v16bf a, v16bf b, v8f c) {
  return __builtin_amdgcn_wmma_f32_16x16x32_bf16(false, a, false, b, (short)0, c,
                                                 false, false);
}

// Stage a [K x Ncols] f32 weight matrix into LDS as bf16, pre-swizzled into
// WMMA B-fragment order: frag(kstep,ntile) is 1024B; within it lane*32B holds
// the lane's 8 VGPRs (2 bf16 each). Lane = ((k>>4)&1)*16 + (n&15), vgpr=(k&15)>>1.
static __device__ __forceinline__ void stage_weight(const float* __restrict__ W,
                                                    unsigned short* lds, int K,
                                                    int Ncols, int tid, int nthr) {
  const int total = K * Ncols;
  for (int i = tid; i < total; i += nthr) {
    int k = i / Ncols, n = i - k * Ncols;
    int lane = (((k >> 4) & 1) << 4) | (n & 15);
    int frag = (k >> 5) * (Ncols >> 4) + (n >> 4);
    int idx  = frag * 512 + lane * 16 + ((k & 15) >> 1) * 2 + (k & 1);
    lds[idx] = f2bf(W[i]);
  }
}

static __device__ __forceinline__ v16bf load_frag_b(const unsigned short* lds,
                                                    int frag, int lane) {
  const v4u* p = (const v4u*)(lds + frag * 512 + lane * 16);
  v4u a = p[0], b = p[1];
  v8u r; r[0]=a[0]; r[1]=a[1]; r[2]=a[2]; r[3]=a[3];
         r[4]=b[0]; r[5]=b[1]; r[6]=b[2]; r[7]=b[3];
  return __builtin_bit_cast(v16bf, r);
}

// A fragment (16 rows x 32 K, bf16) from row-major LDS (rowStride in halfwords).
static __device__ __forceinline__ v16bf load_frag_a(const unsigned short* base,
                                                    int rowStride, int kbase,
                                                    int lane) {
  const unsigned short* p = base + (lane & 15) * rowStride + kbase + (lane >> 4) * 16;
  v4u a = *(const v4u*)p;
  v4u b = *(const v4u*)(p + 8);
  v8u r; r[0]=a[0]; r[1]=a[1]; r[2]=a[2]; r[3]=a[3];
         r[4]=b[0]; r[5]=b[1]; r[6]=b[2]; r[7]=b[3];
  return __builtin_bit_cast(v16bf, r);
}

// Software-pipelined [16 x 320] @ [320 x 128] GEMM: prefetch next B (and next A
// at k-step boundaries) before each WMMA so ds_load latency overlaps the matrix
// pipe instead of a full s_wait_dscnt 0 per WMMA.
static __device__ __forceinline__ void gemm_k320_n128(const unsigned short* aLds,
                                                      const unsigned short* wLds,
                                                      int lane, v8f acc[8]) {
  v16bf aF = load_frag_a(aLds, CAT, 0, lane);
  v16bf bF = load_frag_b(wLds, 0, lane);
#pragma unroll
  for (int ks = 0; ks < 10; ++ks) {
#pragma unroll
    for (int n = 0; n < 8; ++n) {
      v16bf bN = bF, aN = aF;
      if (!(ks == 9 && n == 7))
        bN = load_frag_b(wLds, (n == 7) ? (ks + 1) * 8 : ks * 8 + n + 1, lane);
      if (n == 7 && ks < 9)
        aN = load_frag_a(aLds, CAT, (ks + 1) * 32, lane);
      acc[n] = wmma_bf16(aF, bF, acc[n]);
      bF = bN; aF = aN;
    }
  }
}

// Stage 16 cat rows [N2 | e | N1] as bf16 into wave-private LDS.
static __device__ __forceinline__ void stage_cat(unsigned short* myCat,
                                                 const float* __restrict__ x,
                                                 const float* __restrict__ eAttr,
                                                 const int* __restrict__ eSrc,
                                                 const int* __restrict__ eDst,
                                                 int eBase, int lane) {
  for (int chunk = lane; chunk < 16 * 80; chunk += 32) {
    int m = chunk / 80, seg = chunk - m * 80;   // seg = float4 index in cat row
    int e = eBase + m;
    v4f v = {0.f, 0.f, 0.f, 0.f};
    if (e < NE) {
      if (seg < 32)      v = *(const v4f*)(x + (long)eDst[e] * VI + seg * 4);
      else if (seg < 48) v = *(const v4f*)(eAttr + (long)e * EI + (seg - 32) * 4);
      else               v = *(const v4f*)(x + (long)eSrc[e] * VI + (seg - 48) * 4);
    }
    unsigned* d = (unsigned*)(myCat + m * CAT + seg * 4);
    d[0] = (unsigned)f2bf(v[0]) | ((unsigned)f2bf(v[1]) << 16);
    d[1] = (unsigned)f2bf(v[2]) | ((unsigned)f2bf(v[3]) << 16);
  }
}

static __device__ __forceinline__ void wave_lds_sync() {
  asm volatile("s_wait_dscnt 0" ::: "memory");
  __builtin_amdgcn_wave_barrier();
}

// ---------------------------------------------------------------------------
__global__ void init_ws(float* segMax, float* numAcc, float* denAcc) {
  int i = blockIdx.x * blockDim.x + threadIdx.x;
  if (i < NN * VO) {
    segMax[i] = -__builtin_inff();
    numAcc[i] = 0.f;
    denAcc[i] = 0.f;
  }
}

// ---------------------------------------------------------------------------
// Pass 1: logits = prelu(cat @ W_a); spill bf16 logits; atomic segmax per dst.
__global__ __launch_bounds__(256)
void pass1_logits(const float* __restrict__ x, const int* __restrict__ eSrc,
                  const int* __restrict__ eDst, const float* __restrict__ eAttr,
                  const float* __restrict__ W_a, const float* __restrict__ aPrelu,
                  unsigned short* __restrict__ logits, float* __restrict__ segMax) {
  extern __shared__ __align__(16) unsigned char smem[];
  unsigned short* wA  = (unsigned short*)smem;                 // 320*128 bf16 = 80 KB
  unsigned short* cat = (unsigned short*)(smem + 81920);       // 8 waves * 16*320 bf16

  const int tid = threadIdx.x, lane = tid & 31, wave = tid >> 5;
  const int half = lane >> 4, col0 = lane & 15;
  unsigned short* myCat = cat + wave * (16 * CAT);

  stage_weight(W_a, wA, CAT, VO, tid, blockDim.x);
  __syncthreads();

  const float slope = aPrelu[0];
  const int numTiles = (NE + 127) / 128;

  for (int tile = blockIdx.x; tile < numTiles; tile += gridDim.x) {
    const int eBase = tile * 128 + wave * 16;
    stage_cat(myCat, x, eAttr, eSrc, eDst, eBase, lane);
    wave_lds_sync();

    int myDst[8];
#pragma unroll
    for (int r = 0; r < 8; ++r) {
      int e = eBase + r + half * 8;
      myDst[r] = (e < NE) ? eDst[e] : -1;
    }

    v8f acc[8] = {};
    gemm_k320_n128(myCat, wA, lane, acc);

#pragma unroll
    for (int n = 0; n < 8; ++n) {
#pragma unroll
      for (int r = 0; r < 8; ++r) {
        int dst = myDst[r];
        if (dst < 0) continue;
        float v = acc[n][r];
        v = (v >= 0.f) ? v : slope * v;                  // PReLU
        int e = eBase + r + half * 8, col = n * 16 + col0;
        logits[(long)e * VO + col] = f2bf(v);
        float* mp = segMax + (long)dst * VO + col;       // float max via int trick
        if (v >= 0.f) atomicMax((int*)mp, __float_as_int(v));
        else          atomicMin((unsigned*)mp, (unsigned)__float_as_int(v));
      }
    }
    __builtin_amdgcn_wave_barrier();   // LDS ops are in-order per wave
  }
}

// ---------------------------------------------------------------------------
// Pass 2: messages = cat @ W_T (kept in regs), ex = exp(logit - segmax[dst]),
// num += ex*msg, den += ex (atomics); new_e = (N1+N2)@W_e + e@W_ee stored out.
__global__ __launch_bounds__(256)
void pass2_messages(const float* __restrict__ x, const int* __restrict__ eSrc,
                    const int* __restrict__ eDst, const float* __restrict__ eAttr,
                    const float* __restrict__ W_T, const float* __restrict__ W_e,
                    const float* __restrict__ W_ee,
                    const unsigned short* __restrict__ logits,
                    const float* __restrict__ segMax,
                    float* __restrict__ numAcc, float* __restrict__ denAcc,
                    float* __restrict__ newE) {
  extern __shared__ __align__(16) unsigned char smem[];
  unsigned short* wT   = (unsigned short*)smem;                        // 80 KB
  unsigned short* wE   = (unsigned short*)(smem + 81920);              // 16 KB
  unsigned short* wEE  = (unsigned short*)(smem + 81920 + 16384);      // 8 KB
  unsigned short* cat  = (unsigned short*)(smem + 81920 + 16384 + 8192);
  unsigned short* nsum = cat + 8 * 16 * CAT;                           // 8*16*128 bf16

  const int tid = threadIdx.x, lane = tid & 31, wave = tid >> 5;
  const int half = lane >> 4, col0 = lane & 15;
  unsigned short* myCat  = cat  + wave * (16 * CAT);
  unsigned short* myNsum = nsum + wave * (16 * VI);

  stage_weight(W_T,  wT,  CAT, VO, tid, blockDim.x);
  stage_weight(W_e,  wE,  VI,  EO, tid, blockDim.x);
  stage_weight(W_ee, wEE, EI,  EO, tid, blockDim.x);
  __syncthreads();

  const int numTiles = (NE + 127) / 128;

  for (int tile = blockIdx.x; tile < numTiles; tile += gridDim.x) {
    const int eBase = tile * 128 + wave * 16;
    stage_cat(myCat, x, eAttr, eSrc, eDst, eBase, lane);

    // stage bf16(N1 + N2) for the W_e GEMM (f32 add, then round)
    for (int chunk = lane; chunk < 16 * 32; chunk += 32) {
      int m = chunk >> 5, seg = chunk & 31;
      int e = eBase + m;
      v4f v = {0.f, 0.f, 0.f, 0.f};
      if (e < NE) {
        v4f s1 = *(const v4f*)(x + (long)eSrc[e] * VI + seg * 4);
        v4f s2 = *(const v4f*)(x + (long)eDst[e] * VI + seg * 4);
        v = s1 + s2;
      }
      unsigned* d = (unsigned*)(myNsum + m * VI + seg * 4);
      d[0] = (unsigned)f2bf(v[0]) | ((unsigned)f2bf(v[1]) << 16);
      d[1] = (unsigned)f2bf(v[2]) | ((unsigned)f2bf(v[3]) << 16);
    }
    wave_lds_sync();

    int myDst[8];
#pragma unroll
    for (int r = 0; r < 8; ++r) {
      int e = eBase + r + half * 8;
      myDst[r] = (e < NE) ? eDst[e] : -1;
    }

    // messages GEMM: [16 x 320] @ [320 x 128]
    v8f acc[8] = {};
    gemm_k320_n128(myCat, wT, lane, acc);

    // softmax-weighted scatter into num/den
#pragma unroll
    for (int n = 0; n < 8; ++n) {
#pragma unroll
      for (int r = 0; r < 8; ++r) {
        int dst = myDst[r];
        if (dst < 0) continue;
        int e = eBase + r + half * 8, col = n * 16 + col0;
        float msg = acc[n][r];
        float lg  = bf2f(logits[(long)e * VO + col]);
        float ex  = __expf(lg - segMax[(long)dst * VO + col]);
        atomicAdd(numAcc + (long)dst * VO + col, ex * msg);
        atomicAdd(denAcc + (long)dst * VO + col, ex);
      }
    }

    // new_e GEMM: (N1+N2)[16x128]@W_e[128x64] + e[16x64]@W_ee[64x64]
    v8f acc2[4] = {};
#pragma unroll
    for (int ks = 0; ks < 4; ++ks) {
      v16bf aF = load_frag_a(myNsum, VI, ks * 32, lane);
#pragma unroll
      for (int n = 0; n < 4; ++n)
        acc2[n] = wmma_bf16(aF, load_frag_b(wE, ks * 4 + n, lane), acc2[n]);
    }
#pragma unroll
    for (int ks = 0; ks < 2; ++ks) {
      v16bf aF = load_frag_a(myCat, CAT, 128 + ks * 32, lane);   // e-region of cat
#pragma unroll
      for (int n = 0; n < 4; ++n)
        acc2[n] = wmma_bf16(aF, load_frag_b(wEE, ks * 4 + n, lane), acc2[n]);
    }
#pragma unroll
    for (int n = 0; n < 4; ++n) {
#pragma unroll
      for (int r = 0; r < 8; ++r) {
        int e = eBase + r + half * 8;
        if (e < NE) newE[(long)e * EO + n * 16 + col0] = acc2[n][r];
      }
    }
    __builtin_amdgcn_wave_barrier();
  }
}

// ---------------------------------------------------------------------------
__global__ void pass3_finalize(const float* __restrict__ numAcc,
                               const float* __restrict__ denAcc,
                               const float* __restrict__ b_T,
                               float* __restrict__ newX) {
  int i = blockIdx.x * blockDim.x + threadIdx.x;
  if (i < NN * VO)
    newX[i] = numAcc[i] / (denAcc[i] + 1e-16f) + b_T[i % VO];
}

// ---------------------------------------------------------------------------
extern "C" void kernel_launch(void* const* d_in, const int* in_sizes, int n_in,
                              void* d_out, int out_size, void* d_ws, size_t ws_size,
                              hipStream_t stream) {
  (void)in_sizes; (void)n_in; (void)out_size; (void)ws_size;
  const float* x      = (const float*)d_in[0];
  const int*   eIdx   = (const int*)d_in[1];     // [2, NE] int32
  const float* eAttr  = (const float*)d_in[2];
  const float* W_a    = (const float*)d_in[3];
  const float* W_T    = (const float*)d_in[4];
  const float* b_T    = (const float*)d_in[5];
  const float* W_e    = (const float*)d_in[6];
  const float* W_ee   = (const float*)d_in[7];
  const float* aPrelu = (const float*)d_in[8];

  float* outX = (float*)d_out;
  float* outE = outX + (size_t)NN * VO;

  float* segMax = (float*)d_ws;
  float* numAcc = segMax + (size_t)NN * VO;
  float* denAcc = numAcc + (size_t)NN * VO;
  unsigned short* logits = (unsigned short*)(denAcc + (size_t)NN * VO); // NE*VO bf16

  const int* eSrc = eIdx;
  const int* eDst = eIdx + NE;

  const size_t smem1 = 81920u + 8u * 16u * CAT * 2u;                        // 160 KB
  const size_t smem2 = 81920u + 16384u + 8192u +
                       8u * 16u * CAT * 2u + 8u * 16u * VI * 2u;            // 216 KB
  (void)hipFuncSetAttribute(reinterpret_cast<const void*>(pass1_logits),
                            hipFuncAttributeMaxDynamicSharedMemorySize, (int)smem1);
  (void)hipFuncSetAttribute(reinterpret_cast<const void*>(pass2_messages),
                            hipFuncAttributeMaxDynamicSharedMemorySize, (int)smem2);

  init_ws<<<(NN * VO + 255) / 256, 256, 0, stream>>>(segMax, numAcc, denAcc);
  pass1_logits<<<1024, 256, smem1, stream>>>(x, eSrc, eDst, eAttr, W_a, aPrelu,
                                             logits, segMax);
  pass2_messages<<<1024, 256, smem2, stream>>>(x, eSrc, eDst, eAttr, W_T, W_e, W_ee,
                                               logits, segMax, numAcc, denAcc, outE);
  pass3_finalize<<<(NN * VO + 255) / 256, 256, 0, stream>>>(numAcc, denAcc, b_T, outX);
}
